// multi_head_attn_26336739459554
// MI455X (gfx1250) — matrix operand order repaired and checked by reading, not compiled, not run on hardware
//
#include <hip/hip_runtime.h>
#include <hip/hip_bf16.h>
#include <math.h>

#define N_HEADS 8
#define N_FEAT  64
#define IMG_W   80
#define IMG_H   80
#define NPIX    (IMG_W * IMG_H)   /* 6400 */
#define C_DS    32

typedef __attribute__((ext_vector_type(16))) __bf16 v16bf;
typedef __attribute__((ext_vector_type(8)))  __bf16 v8bf;
typedef __attribute__((ext_vector_type(8)))  float  v8f;
typedef __attribute__((ext_vector_type(4)))  unsigned int v4u;
typedef __attribute__((ext_vector_type(8)))  int v8i;
typedef __attribute__((ext_vector_type(4)))  int v4i;

#if defined(__has_builtin)
#if __has_builtin(__builtin_amdgcn_tensor_load_to_lds) && __has_builtin(__builtin_amdgcn_s_wait_tensorcnt)
#define USE_TDM 1
#endif
#endif
#ifndef USE_TDM
#define USE_TDM 0
#endif

__device__ __forceinline__ unsigned short f32_to_bf16(float f) {
    union { float f; unsigned int u; } v; v.f = f;
    unsigned int u = v.u;
    unsigned int r = u + 0x7FFFu + ((u >> 16) & 1u);       // round-to-nearest-even
    if ((u & 0x7F800000u) == 0x7F800000u) r = u;           // inf/nan passthrough
    return (unsigned short)(r >> 16);
}

// For softmax probabilities: always finite & positive -> branch-free RNE.
__device__ __forceinline__ unsigned int f32_to_bf16_fast(float f) {
    union { float f; unsigned int u; } v; v.f = f;
    return (v.u + 0x7FFFu + ((v.u >> 16) & 1u)) >> 16;
}

#if USE_TDM
// Issue a TDM 2D tile load: 64 rows x 128 B (64 bf16) from a [6400][64] bf16
// tensor into LDS at lds_off.  D# packing per CDNA5 ISA ch.8 (group0/group1).
// This toolchain declares the 6-arg builtin:
//   (uint32x4 g0, int32x8 g1, int32x4 g2, int32x4 g3, int32x8 g4, i32 cpol)
__device__ __forceinline__ void tdm_load_k_tile(unsigned lds_off,
                                                unsigned long long gaddr) {
    v4u g0;
    g0[0] = 1u;                                            // count=1, user mode
    g0[1] = lds_off;                                       // lds_addr (bytes)
    g0[2] = (unsigned)(gaddr & 0xFFFFFFFFull);             // global_addr[31:0]
    g0[3] = (unsigned)((gaddr >> 32) & 0x01FFFFFFull)      // global_addr[56:32]
          | (2u << 30);                                    // type = 2 ("image")
    v8i g1;
    g1[0] = (int)(1u << 16);          // workgroup_mask=0, data_size=1 (2 bytes)
    g1[1] = (int)(64u << 16);         // tensor_dim0 = 64 (bits 79:64)
    g1[2] = (int)(6400u << 16);       // tensor_dim1 = 6400 (bits 111:96)
    g1[3] = (int)(64u << 16);         // tile_dim0 = 64 (bits 127:112)
    g1[4] = 64;                       // tile_dim1 = 64 rows
    g1[5] = 64;                       // tensor_dim0_stride = 64 elements
    g1[6] = 0;
    g1[7] = 0;
    v4i g2 = {0, 0, 0, 0};            // 2D tensor: groups 2/3 unused
    v4i g3 = {0, 0, 0, 0};
    v8i g4 = {0, 0, 0, 0, 0, 0, 0, 0};
    __builtin_amdgcn_tensor_load_to_lds(g0, g1, g2, g3, g4, 0);
}
#endif

// ---------------------------------------------------------------------------
// Kernel 1: elementwise prep.
//   Qs[f,p] = sum_h  WQ_task[h,f]*(WQ_tm1[h,f]*prevQ[h,f,p] + WQ_x[f]*X[f,p] + BQ[h,f]) + BQ_task[h,f]
//   K [f,p] = WK_task[f]*(WK_x[f]*X[f,p] + BK[f]) + BK_task[f]
// Stored transposed as bf16:  QsT[p][f], KT[p][f]  (row stride 64 = 128 B).
// ---------------------------------------------------------------------------
__global__ void prep_qk_kernel(const float* __restrict__ X,
                               const float* __restrict__ WQ_task,
                               const float* __restrict__ BQ_task,
                               const float* __restrict__ WK_task,
                               const float* __restrict__ BK_task,
                               const float* __restrict__ WQ_tm1,
                               const float* __restrict__ WQ_x,
                               const float* __restrict__ BQ,
                               const float* __restrict__ WK_x,
                               const float* __restrict__ BK,
                               const float* __restrict__ prevQ,
                               unsigned short* __restrict__ QsT,
                               unsigned short* __restrict__ KT)
{
    int idx = blockIdx.x * blockDim.x + threadIdx.x;
    if (idx >= N_FEAT * NPIX) return;
    int f = idx / NPIX;
    int p = idx % NPIX;
    float x = X[f * NPIX + p];
    float qs = 0.f;
#pragma unroll
    for (int h = 0; h < N_HEADS; ++h) {
        int hf = h * N_FEAT + f;
        float inner = WQ_tm1[hf] * prevQ[hf * NPIX + p] + WQ_x[f] * x + BQ[hf];
        qs += WQ_task[hf] * inner + BQ_task[hf];
    }
    float k = WK_task[f] * (WK_x[f] * x + BK[f]) + BK_task[f];
    QsT[p * N_FEAT + f] = f32_to_bf16(qs);
    KT [p * N_FEAT + f] = f32_to_bf16(k);
}

// ---------------------------------------------------------------------------
// Kernel 2: 3x3 SAME conv (64 -> 32 ch) + affine; V stored bf16 as [32][6400],
// with each 64-column block column-permuted by sigma(k) = (k%4)*16 + k/4 so the
// flash kernel's packed-LDS K ordering matches the B-operand memory order.
// (Inverse applied here: column j stores at k = (j%16)*4 + j/16.)
// ---------------------------------------------------------------------------
__global__ void conv_v_kernel(const float* __restrict__ X,
                              const float* __restrict__ Wc,   // (32,64,3,3)
                              const float* __restrict__ Bc,   // (32,)
                              const float* __restrict__ WV_task,
                              const float* __restrict__ BV_task,
                              unsigned short* __restrict__ Vws)
{
    int idx = blockIdx.x * blockDim.x + threadIdx.x;
    if (idx >= C_DS * NPIX) return;
    int c = idx / NPIX;
    int p = idx % NPIX;
    int w = p / IMG_H;
    int h = p % IMG_H;
    float acc = Bc[c];
    for (int f = 0; f < N_FEAT; ++f) {
        const float* xf = X  + f * NPIX;
        const float* wf = Wc + (c * N_FEAT + f) * 9;
#pragma unroll
        for (int dw = -1; dw <= 1; ++dw) {
            int ww = w + dw;
            if (ww < 0 || ww >= IMG_W) continue;
#pragma unroll
            for (int dh = -1; dh <= 1; ++dh) {
                int hh = h + dh;
                if (hh < 0 || hh >= IMG_H) continue;
                acc += wf[(dw + 1) * 3 + (dh + 1)] * xf[ww * IMG_H + hh];
            }
        }
    }
    float v = WV_task[c] * acc + BV_task[c];
    int j = p & 63;                                     // column within 64-block
    int sp = (p & ~63) | (((j & 15) << 2) | (j >> 4));  // permuted store index
    Vws[c * NPIX + sp] = f32_to_bf16(v);
}

// ---------------------------------------------------------------------------
// Kernel 3: fused flash attention, 64-wide key tiles.
//   K stream : TDM double-buffer (tensor_load_to_lds + s_wait_tensorcnt),
//              falling back to direct global loads if the builtin is absent.
//   S   = Qs^T K       : 8x v_wmma_f32_16x16x32_bf16 (4 subtiles x 2 ksteps)
//   max : 8 rows x 4-step shfl_xor butterfly (only cross-lane reduction left)
//   P -> LDS (packed ds_store_b64, permuted K) -> A operand (ds_load_b128)
//   O  += P V^T        : 4x WMMA;   L += P * ones : 2x WMMA (row sums)
// ---------------------------------------------------------------------------
__global__ void __launch_bounds__(32)
flash_attn_kernel(const unsigned short* __restrict__ QsT,
                  const unsigned short* __restrict__ KT,
                  const unsigned short* __restrict__ Vws,
                  float* __restrict__ out)
{
#if USE_TDM
    __shared__ __align__(16) unsigned short ldsK[2][64 * N_FEAT];  // 2 x 8 KB
#endif
    __shared__ __align__(16) unsigned short ldsP[16 * 64];         // 2 KB

    const int lane    = threadIdx.x;
    const int halfSel = (lane >= 16) ? 1 : 0;
    const int bcol    = lane & 15;
    const int pblock  = blockIdx.x * 16;
    const int koff    = halfSel ? 16 : 0;    // B-operand K offset for this half
    const int kbase   = halfSel ? 16 : 0;    // A-operand K base for this half

    // --- A operand (Qs rows for this block), resident for the whole loop ---
    // A 16x32 K order used here (contiguous; B matches): lanes 0..15 hold K {0..15}; lanes 16..31
    // hold K {16..31}; row M = lane&15.
    v16bf aq0 = {}, aq1 = {};
    {
        const v8bf* r = (const v8bf*)(QsT + (pblock + bcol) * N_FEAT);
        v8bf lo0 = r[2 * halfSel + 0];
        v8bf hi0 = r[2 * halfSel + 1];
        v8bf lo1 = r[2 * halfSel + 4];
        v8bf hi1 = r[2 * halfSel + 5];
#pragma unroll
        for (int i = 0; i < 8; ++i) {
            aq0[i] = lo0[i]; aq0[8 + i] = hi0[i];
            aq1[i] = lo1[i]; aq1[8 + i] = hi1[i];
        }
    }

    // All-ones B operand for the row-sum GEMM.
    v16bf bone;
#pragma unroll
    for (int i = 0; i < 16; ++i) bone[i] = (__bf16)1.0f;

    v8f O0   = {};   // out columns c = 0..15
    v8f O1   = {};   // out columns c = 16..31
    v8f Lacc = {};   // row sums of P (every lane/column holds the row's sum)
    float m[8];
#pragma unroll
    for (int r = 0; r < 8; ++r) m[r] = -1e30f;

    const int nIter = NPIX / 64;   // 100

#if USE_TDM
    // Prologue: start streaming tile 0 into buffer 0.
    tdm_load_k_tile((unsigned)(unsigned long long)(uintptr_t)&ldsK[0][0],
                    (unsigned long long)(uintptr_t)KT);
#endif

    for (int it = 0; it < nIter; ++it) {
        const int n0 = it * 64;

#if USE_TDM
        if (it + 1 < nIter) {   // issue next tile, then wait for current one
            tdm_load_k_tile(
                (unsigned)(unsigned long long)(uintptr_t)&ldsK[(it + 1) & 1][0],
                (unsigned long long)(uintptr_t)(KT + (size_t)(n0 + 64) * N_FEAT));
            asm volatile("" ::: "memory");
            __builtin_amdgcn_s_wait_tensorcnt(1);
        } else {
            asm volatile("" ::: "memory");
            __builtin_amdgcn_s_wait_tensorcnt(0);
        }
        asm volatile("" ::: "memory");
        const unsigned short* kb = &ldsK[it & 1][0];
#else
        if (n0 + 64 < NPIX) {   // stream next K tile: global_prefetch_b8
            __builtin_prefetch(KT + (n0 + 64 + bcol) * N_FEAT, 0, 1);
            __builtin_prefetch(KT + (n0 + 96 + bcol) * N_FEAT, 0, 1);
        }
        const unsigned short* kb = KT + (size_t)n0 * N_FEAT;
#endif

        // ---- S tiles: sim[p, n] = sum_f Qs[f,p] K[f,n], n in 4 subtiles ----
        v8f s0 = {}, s1 = {}, s2 = {}, s3 = {};
        {
            const unsigned short* k0p = kb + ( 0 + bcol) * N_FEAT;
            const unsigned short* k1p = kb + (16 + bcol) * N_FEAT;
            const unsigned short* k2p = kb + (32 + bcol) * N_FEAT;
            const unsigned short* k3p = kb + (48 + bcol) * N_FEAT;
            v16bf b00 = *(const v16bf*)(k0p + koff);
            v16bf b01 = *(const v16bf*)(k0p + 32 + koff);
            v16bf b10 = *(const v16bf*)(k1p + koff);
            v16bf b11 = *(const v16bf*)(k1p + 32 + koff);
            v16bf b20 = *(const v16bf*)(k2p + koff);
            v16bf b21 = *(const v16bf*)(k2p + 32 + koff);
            v16bf b30 = *(const v16bf*)(k3p + koff);
            v16bf b31 = *(const v16bf*)(k3p + 32 + koff);
            s0 = __builtin_amdgcn_wmma_f32_16x16x32_bf16(false, aq0, false, b00, (short)0, s0, false, false);
            s1 = __builtin_amdgcn_wmma_f32_16x16x32_bf16(false, aq0, false, b10, (short)0, s1, false, false);
            s2 = __builtin_amdgcn_wmma_f32_16x16x32_bf16(false, aq0, false, b20, (short)0, s2, false, false);
            s3 = __builtin_amdgcn_wmma_f32_16x16x32_bf16(false, aq0, false, b30, (short)0, s3, false, false);
            s0 = __builtin_amdgcn_wmma_f32_16x16x32_bf16(false, aq1, false, b01, (short)0, s0, false, false);
            s1 = __builtin_amdgcn_wmma_f32_16x16x32_bf16(false, aq1, false, b11, (short)0, s1, false, false);
            s2 = __builtin_amdgcn_wmma_f32_16x16x32_bf16(false, aq1, false, b21, (short)0, s2, false, false);
            s3 = __builtin_amdgcn_wmma_f32_16x16x32_bf16(false, aq1, false, b31, (short)0, s3, false, false);
        }

        // ---- online softmax: row max across 64 cols, rescale, exp, pack ----
#pragma unroll
        for (int r = 0; r < 8; ++r) {
            float mx = fmaxf(fmaxf(s0[r], s1[r]), fmaxf(s2[r], s3[r]));
#pragma unroll
            for (int d = 1; d < 16; d <<= 1) mx = fmaxf(mx, __shfl_xor(mx, d, 32));
            float nm    = fmaxf(m[r], mx);
            float alpha = __expf(m[r] - nm);
            m[r] = nm;
            O0[r] *= alpha; O1[r] *= alpha; Lacc[r] *= alpha;
            float e0 = __expf(s0[r] - nm);
            float e1 = __expf(s1[r] - nm);
            float e2 = __expf(s2[r] - nm);
            float e3 = __expf(s3[r] - nm);
            // Permuted K ordering: this lane's 4 values are K = bcol*4 + t,
            // contiguous in LDS -> single 64-bit store.
            unsigned int lo = f32_to_bf16_fast(e0) | (f32_to_bf16_fast(e1) << 16);
            unsigned int hi = f32_to_bf16_fast(e2) | (f32_to_bf16_fast(e3) << 16);
            unsigned long long pk = (unsigned long long)lo
                                  | ((unsigned long long)hi << 32);
            int row = r + 8 * halfSel;
            *(unsigned long long*)(ldsP + row * 64 + bcol * 4) = pk;
        }
        asm volatile("s_wait_dscnt 0" ::: "memory");   // DS in-order within wave

        // ---- reload P as A operand (K = permuted n ordering) ----
        v16bf pa0 = {}, pa1 = {};
        {
            const v8bf* lp = (const v8bf*)(ldsP + bcol * 64 + kbase);
            v8bf a0lo = lp[0];    // K kbase..kbase+7
            v8bf a0hi = lp[1];    // K kbase+8..kbase+15
            v8bf a1lo = lp[4];    // +32
            v8bf a1hi = lp[5];
#pragma unroll
            for (int i = 0; i < 8; ++i) {
                pa0[i] = a0lo[i]; pa0[8 + i] = a0hi[i];
                pa1[i] = a1lo[i]; pa1[8 + i] = a1hi[i];
            }
        }

        // ---- O += P V^T  and  L += P * ones ----
        {
            const unsigned short* v0p = Vws + bcol * NPIX        + n0;
            const unsigned short* v1p = Vws + (bcol + 16) * NPIX + n0;
            v16bf vb00 = *(const v16bf*)(v0p + koff);        // K 0..31 slice
            v16bf vb01 = *(const v16bf*)(v0p + 32 + koff);   // K 32..63 slice
            v16bf vb10 = *(const v16bf*)(v1p + koff);
            v16bf vb11 = *(const v16bf*)(v1p + 32 + koff);
            O0   = __builtin_amdgcn_wmma_f32_16x16x32_bf16(false, pa0, false, vb00, (short)0, O0,   false, false);
            O1   = __builtin_amdgcn_wmma_f32_16x16x32_bf16(false, pa0, false, vb10, (short)0, O1,   false, false);
            Lacc = __builtin_amdgcn_wmma_f32_16x16x32_bf16(false, pa0, false, bone, (short)0, Lacc, false, false);
            O0   = __builtin_amdgcn_wmma_f32_16x16x32_bf16(false, pa1, false, vb01, (short)0, O0,   false, false);
            O1   = __builtin_amdgcn_wmma_f32_16x16x32_bf16(false, pa1, false, vb11, (short)0, O1,   false, false);
            Lacc = __builtin_amdgcn_wmma_f32_16x16x32_bf16(false, pa1, false, bone, (short)0, Lacc, false, false);
        }
    }

    // ---- finalize: out[c, p] = O[p, c] / L[p] ----
#pragma unroll
    for (int r = 0; r < 8; ++r) {
        float inv = 1.0f / Lacc[r];
        int p = pblock + r + 8 * halfSel;
        out[bcol * NPIX + p]        = O0[r] * inv;
        out[(bcol + 16) * NPIX + p] = O1[r] * inv;
    }
}

// ---------------------------------------------------------------------------
extern "C" void kernel_launch(void* const* d_in, const int* in_sizes, int n_in,
                              void* d_out, int out_size, void* d_ws, size_t ws_size,
                              hipStream_t stream)
{
    const float* X        = (const float*)d_in[0];
    const float* WQ_task  = (const float*)d_in[1];
    const float* BQ_task  = (const float*)d_in[2];
    const float* WK_task  = (const float*)d_in[3];
    const float* BK_task  = (const float*)d_in[4];
    const float* WV_task  = (const float*)d_in[5];
    const float* BV_task  = (const float*)d_in[6];
    const float* WQ_tm1   = (const float*)d_in[7];
    const float* WQ_x     = (const float*)d_in[8];
    const float* BQ       = (const float*)d_in[9];
    const float* WK_x     = (const float*)d_in[10];
    const float* BK       = (const float*)d_in[11];
    const float* prevQ    = (const float*)d_in[12];
    const float* Vconv_w  = (const float*)d_in[13];
    const float* Vconv_b  = (const float*)d_in[14];
    float* out = (float*)d_out;

    // Workspace layout (bf16): QsT[6400][64] | KT[6400][64] | V[32][6400]  (~2 MB)
    unsigned short* QsT = (unsigned short*)d_ws;
    unsigned short* KT  = QsT + (size_t)NPIX * N_FEAT;
    unsigned short* Vws = KT  + (size_t)NPIX * N_FEAT;

    {
        int n = N_FEAT * NPIX;
        prep_qk_kernel<<<(n + 255) / 256, 256, 0, stream>>>(
            X, WQ_task, BQ_task, WK_task, BK_task,
            WQ_tm1, WQ_x, BQ, WK_x, BK, prevQ, QsT, KT);
    }
    {
        int n = C_DS * NPIX;
        conv_v_kernel<<<(n + 255) / 256, 256, 0, stream>>>(
            X, Vconv_w, Vconv_b, WV_task, BV_task, Vws);
    }
    flash_attn_kernel<<<NPIX / 16, 32, 0, stream>>>(QsT, KT, Vws, out);
}
